// NSGNNLayer_89309549953493
// MI455X (gfx1250) — compile-verified, matrix operand order checked by reading, not run on hardware
//
#include <hip/hip_runtime.h>

#define D_DIM 128

typedef __attribute__((ext_vector_type(16))) __bf16 v16bf;
typedef __attribute__((ext_vector_type(8)))  float  v8f;

__device__ __forceinline__ void atomic_add_f32(float* p, float v) {
    __hip_atomic_fetch_add(p, v, __ATOMIC_RELAXED, __HIP_MEMORY_SCOPE_AGENT);
}

// ---------------------------------------------------------------------------
// init: deg[i] = 1.0 (self-loop weight), zero the BN stat accumulators
// ---------------------------------------------------------------------------
__global__ void k_init(float* __restrict__ deg, float* __restrict__ stats, int N) {
    int i = blockIdx.x * blockDim.x + threadIdx.x;
    if (i < N)   deg[i] = 1.0f;
    if (i < 512) stats[i] = 0.0f;
}

// ---------------------------------------------------------------------------
// degree accumulation over edges: deg[col] += w[e]
// ---------------------------------------------------------------------------
__global__ void k_deg_edges(const long long* __restrict__ ei,
                            const float* __restrict__ ew,
                            float* __restrict__ deg, int E) {
    int e = blockIdx.x * blockDim.x + threadIdx.x;
    if (e >= E) return;
    long long c = ei[(size_t)E + e];
    atomic_add_f32(&deg[c], ew[e]);
}

// deg -> dis = rsqrt(deg)  (deg >= 1 always due to self-loop)
__global__ void k_make_dis(float* __restrict__ deg, int N) {
    int i = blockIdx.x * blockDim.x + threadIdx.x;
    if (i < N) deg[i] = rsqrtf(deg[i]);
}

// ---------------------------------------------------------------------------
// WMMA GEMM: out[M,NCOLS] = A[M,K] @ W[K,NCOLS] (+bias) (relu?) (+residual)
//
// W is staged in LDS once per block, converted to bf16 and pre-swizzled into
// B-fragment layout: slot(kstep, tileN, lane) is a 32B chunk holding the 16
// bf16 values that lane needs -> B frag = one 32B LDS read (2x ds_load_b128)
// instead of 16 strided global_load_b32.
//
// Fragment layouts per ISA 7.12.2 (wave32):
//   A/B: lane&15 = M (A) / N (B); koff = (lane&16)?8:0;
//        frag elem j -> K = k0 + koff + (j&7) + 16*(j>>3)
//   inverse (for the pack): kin=k&31 -> j=(kin&7)+((kin&16)?8:0),
//        lane_hi = (kin&8)?16:0
//   C/D: vgpr r -> row r + ((lane&16)?8:0), col = lane&15
//
// Each wave computes NT=2 adjacent 16x16 tiles sharing one A frag.
// 8 waves / 256-thread block:
//   NCOLS=128: 4 waves per 16-row strip, 2 strips -> block = 32 rows
//   NCOLS=256: 8 waves per strip, 1 strip         -> block = 16 rows
// ---------------------------------------------------------------------------
template<int K, int NCOLS, bool BIAS, bool RELU, bool RES>
__global__ void __launch_bounds__(256)
k_gemm_wmma(const float* __restrict__ A, const float* __restrict__ W,
            const float* __restrict__ bias, const float* __restrict__ residual,
            float* __restrict__ out, int M) {
    constexpr int NT      = 2;                    // tiles per wave
    constexpr int WPS     = NCOLS / (16 * NT);    // waves per row strip
    constexpr int STRIPS  = 8 / WPS;              // row strips per block
    constexpr int KSTEPS  = K / 32;
    constexpr int NTILES  = NCOLS / 16;

    __shared__ __align__(16) char ldsb[(size_t)K * NCOLS * 2];   // bf16 W frags

    const int tid = threadIdx.x;

    // ---- pack W -> LDS (coalesced float4 reads, each element read once) ----
#pragma unroll 1
    for (int e4 = tid * 4; e4 < K * NCOLS; e4 += 256 * 4) {
        const int k = e4 / NCOLS;
        const int n = e4 % NCOLS;                 // n..n+3 within one row
        const float4 w4 = *(const float4*)(W + (size_t)k * NCOLS + n);
        const int kin = k & 31, ks = k >> 5;
        const int lhi = (kin & 8) ? 16 : 0;
        const int j   = (kin & 7) + ((kin & 16) ? 8 : 0);
        const float vals[4] = {w4.x, w4.y, w4.z, w4.w};
#pragma unroll
        for (int q = 0; q < 4; ++q) {
            const int nn   = n + q;
            const int lane = lhi | (nn & 15);
            const int tile = nn >> 4;
            const int off  = (((ks * NTILES) + tile) * 32 + lane) * 32 + j * 2;
            *(__bf16*)(ldsb + off) = (__bf16)vals[q];
        }
    }
    __syncthreads();

    const int lane  = tid & 31;
    const int wave  = tid >> 5;
    const int strip = wave / WPS;
    const int wcol  = wave % WPS;

    const int row0 = (blockIdx.x * STRIPS + strip) * 16;
    if (row0 < M) {                               // wave-uniform; after barrier
        const int idx  = lane & 15;
        const int koff = (lane & 16) ? 8 : 0;

        v8f acc[NT] = {};
#pragma unroll
        for (int ks = 0; ks < KSTEPS; ++ks) {
            // ---- A fragment: 4x b128 (32B aligned: koff in {0,8}) ----
            v16bf a;
            const float4* a4 =
                (const float4*)(A + (size_t)(row0 + idx) * K + (ks * 32 + koff));
            const float4 p0 = a4[0], p1 = a4[1];  // K = koff+0..7
            const float4 p2 = a4[4], p3 = a4[5];  // K = koff+16..23
            a[0]  = (__bf16)p0.x; a[1]  = (__bf16)p0.y; a[2]  = (__bf16)p0.z; a[3]  = (__bf16)p0.w;
            a[4]  = (__bf16)p1.x; a[5]  = (__bf16)p1.y; a[6]  = (__bf16)p1.z; a[7]  = (__bf16)p1.w;
            a[8]  = (__bf16)p2.x; a[9]  = (__bf16)p2.y; a[10] = (__bf16)p2.z; a[11] = (__bf16)p2.w;
            a[12] = (__bf16)p3.x; a[13] = (__bf16)p3.y; a[14] = (__bf16)p3.z; a[15] = (__bf16)p3.w;

#pragma unroll
            for (int t = 0; t < NT; ++t) {
                const int tileN = wcol * NT + t;
                const v16bf b = *(const v16bf*)(
                    ldsb + (((ks * NTILES) + tileN) * 32 + lane) * 32);
                acc[t] = __builtin_amdgcn_wmma_f32_16x16x32_bf16(
                    false, a, false, b, (short)0, acc[t], false, false);
            }
        }

        const int rbase = (lane & 16) ? 8 : 0;
#pragma unroll
        for (int t = 0; t < NT; ++t) {
            const int col = (wcol * NT + t) * 16 + idx;
            const float bv = BIAS ? bias[col] : 0.0f;
#pragma unroll
            for (int r = 0; r < 8; ++r) {
                const int row = row0 + rbase + r;
                float v = acc[t][r] + bv;
                if (RELU) v = fmaxf(v, 0.0f);
                if (RES)  v += residual[(size_t)row * NCOLS + col];
                out[(size_t)row * NCOLS + col] = v;
            }
        }
    }
}

// ---------------------------------------------------------------------------
// h = x + b_gcn + dis^2 * xw   (residual + GCN bias + self-loop message)
// ---------------------------------------------------------------------------
__global__ void k_h_init(const float* __restrict__ x, const float* __restrict__ b_gcn,
                         const float* __restrict__ dis, const float* __restrict__ xw,
                         float* __restrict__ h, long long total) {
    long long idx = (long long)blockIdx.x * blockDim.x + threadIdx.x;
    if (idx >= total) return;
    int i = (int)(idx >> 7);
    int d = (int)(idx & 127);
    float dd = dis[i];
    h[idx] = x[idx] + b_gcn[d] + dd * dd * xw[idx];
}

// ---------------------------------------------------------------------------
// edge scatter: one wave per edge; lane handles 4 contiguous floats (float4)
// h[col] += dis[row]*w*dis[col] * xw[row]
// ---------------------------------------------------------------------------
__global__ void __launch_bounds__(256)
k_edge_scatter(const long long* __restrict__ ei, const float* __restrict__ ew,
               const float* __restrict__ dis, const float* __restrict__ xw,
               float* __restrict__ h, int E) {
    const int lane = threadIdx.x & 31;
    const int wpb  = blockDim.x >> 5;
    const long long e = (long long)blockIdx.x * wpb + (threadIdx.x >> 5);
    if (e >= E) return;                       // wave-uniform
    const long long r = ei[e];
    const long long c = ei[(size_t)E + e];
    const float norm = dis[r] * ew[e] * dis[c];
    const float4 v = ((const float4*)(xw + (size_t)r * D_DIM))[lane];
    float* dst = h + (size_t)c * D_DIM + lane * 4;
    atomic_add_f32(dst + 0, norm * v.x);
    atomic_add_f32(dst + 1, norm * v.y);
    atomic_add_f32(dst + 2, norm * v.z);
    atomic_add_f32(dst + 3, norm * v.w);
}

// ---------------------------------------------------------------------------
// BN column stats: per-thread register accumulation, then atomic into [128]
// ---------------------------------------------------------------------------
__global__ void k_bn_stats(const float* __restrict__ h,
                           float* __restrict__ sum, float* __restrict__ sumsq, int N) {
    const int d    = threadIdx.x & 127;
    const int sub  = threadIdx.x >> 7;               // 0..1 (256 threads)
    const int step = (blockDim.x >> 7) * gridDim.x;
    float s = 0.0f, ss = 0.0f;
    for (int i = blockIdx.x * (blockDim.x >> 7) + sub; i < N; i += step) {
        float v = h[(size_t)i * D_DIM + d];
        s += v; ss += v * v;
    }
    atomic_add_f32(&sum[d], s);
    atomic_add_f32(&sumsq[d], ss);
}

// BN apply: out = (h - m) * rsqrt(var + eps) * g + b
__global__ void k_bn_apply(const float* __restrict__ h,
                           const float* __restrict__ sum, const float* __restrict__ sumsq,
                           const float* __restrict__ g, const float* __restrict__ b,
                           float* __restrict__ out, int N, long long total) {
    long long idx = (long long)blockIdx.x * blockDim.x + threadIdx.x;
    if (idx >= total) return;
    int d = (int)(idx & 127);
    float invN = 1.0f / (float)N;
    float m   = sum[d] * invN;
    float var = sumsq[d] * invN - m * m;
    float sc  = rsqrtf(var + 1e-5f) * g[d];
    out[idx] = (h[idx] - m) * sc + b[d];
}

// ---------------------------------------------------------------------------
extern "C" void kernel_launch(void* const* d_in, const int* in_sizes, int n_in,
                              void* d_out, int out_size, void* d_ws, size_t ws_size,
                              hipStream_t stream) {
    const float*     x     = (const float*)d_in[0];
    const long long* ei    = (const long long*)d_in[1];
    const float*     ew    = (const float*)d_in[2];
    const float*     W_gcn = (const float*)d_in[3];
    const float*     b_gcn = (const float*)d_in[4];
    const float*     bn1_g = (const float*)d_in[5];
    const float*     bn1_b = (const float*)d_in[6];
    const float*     W_ff1 = (const float*)d_in[7];
    const float*     b_ff1 = (const float*)d_in[8];
    const float*     W_ff2 = (const float*)d_in[9];
    const float*     b_ff2 = (const float*)d_in[10];
    const float*     bn2_g = (const float*)d_in[11];
    const float*     bn2_b = (const float*)d_in[12];
    float* out = (float*)d_out;

    const int N = in_sizes[0] / D_DIM;
    const int E = in_sizes[1] / 2;
    const long long ND = (long long)N * D_DIM;

    float* ws   = (float*)d_ws;
    float* xw   = ws;                  // N*D  (reused as hbn after BN1)
    float* h    = ws + ND;             // N*D  (reused as h2 after FFN)
    float* ff1  = ws + 2 * ND;         // N*2D
    float* deg  = ws + 4 * ND;         // N    (becomes dis)
    float* st   = ws + 4 * ND + N;     // 512
    float* sum1 = st, *sq1 = st + 128, *sum2 = st + 256, *sq2 = st + 384;

    const int B = 256;
    const int gElem = (int)((ND + B - 1) / B);

    // 1) init deg + stats
    {
        int n = N > 512 ? N : 512;
        k_init<<<(n + B - 1) / B, B, 0, stream>>>(deg, st, N);
    }
    // 2) degree over edges
    k_deg_edges<<<(E + B - 1) / B, B, 0, stream>>>(ei, ew, deg, E);
    // 3) dis = rsqrt(deg)
    k_make_dis<<<(N + B - 1) / B, B, 0, stream>>>(deg, N);
    // 4) xw = x @ W_gcn        (K=128, NCOLS=128; block covers 32 rows)
    k_gemm_wmma<128, 128, false, false, false>
        <<<(N + 31) / 32, B, 0, stream>>>(x, W_gcn, nullptr, nullptr, xw, N);
    // 5) h = x + b_gcn + dis^2 * xw
    k_h_init<<<gElem, B, 0, stream>>>(x, b_gcn, deg, xw, h, ND);
    // 6) edge scatter (one wave per edge, 8 waves/block)
    k_edge_scatter<<<(E + 7) / 8, B, 0, stream>>>(ei, ew, deg, xw, h, E);
    // 7) BN1 stats
    k_bn_stats<<<256, B, 0, stream>>>(h, sum1, sq1, N);
    // 8) BN1 apply -> hbn (reuse xw buffer)
    k_bn_apply<<<gElem, B, 0, stream>>>(h, sum1, sq1, bn1_g, bn1_b, xw, N, ND);
    // 9) ff1 = relu(hbn @ W_ff1 + b_ff1)   (K=128, NCOLS=256; block = 16 rows)
    k_gemm_wmma<128, 256, true, true, false>
        <<<(N + 15) / 16, B, 0, stream>>>(xw, W_ff1, b_ff1, nullptr, ff1, N);
    // 10) h2 = hbn + ff1 @ W_ff2 + b_ff2   (K=256, NCOLS=128) -> reuse h buffer
    k_gemm_wmma<256, 128, true, false, true>
        <<<(N + 31) / 32, B, 0, stream>>>(ff1, W_ff2, b_ff2, xw, h, N);
    // 11) BN2 stats
    k_bn_stats<<<256, B, 0, stream>>>(h, sum2, sq2, N);
    // 12) BN2 apply -> out
    k_bn_apply<<<gElem, B, 0, stream>>>(h, sum2, sq2, bn2_g, bn2_b, out, N, ND);
}
